// MultiHeadAttention_49374944035457
// MI455X (gfx1250) — matrix operand order repaired and checked by reading, not compiled, not run on hardware
//
#include <hip/hip_runtime.h>
#include <hip/hip_bf16.h>

// MHA forward for MI455X (gfx1250, wave32, WMMA f16 16x16x32).
//   B=4, S=2048, D=1024, H=16, DK=64, MAX_LEN=2048
//
// Pipeline:
//   0) f32->f16 pre-conversion of q/k/v inputs + all four weights (loads in
//      the GEMM hot loops are pure f16 b128, no cvt on the WMMA path), and
//      rel-bias table transpose (4095,16)->(16,4095) for coalesced gathers.
//   1) qkv_proj: X @ W^T + b, 32x64 tile/wave, software-pipelined (all of
//      step k+32's fragment loads issued before step k's 8 WMMAs).
//      Q,K stored (B,H,S,DK) f16; V stored transposed (B,H,DK,S) f16.
//   2) flash_attn: streaming softmax, 16-query tile/wave, 32 keys/step;
//      K fragments double-buffered across steps, V fragment loads issued
//      early so they overlap the softmax VALU + LDS P relayout.
//   3) out_proj: ctx @ Wo^T + bo -> f32, same pipelined 32x64 tiling.
//
// Workspace regions are padded so the one-iteration-ahead prefetch never
// reads outside d_ws.

typedef __attribute__((ext_vector_type(16))) _Float16 v16h;
typedef __attribute__((ext_vector_type(8)))  _Float16 v8h;
typedef __attribute__((ext_vector_type(8)))  float    v8f;

#define BATCH  4
#define SEQ    2048
#define DMODEL 1024
#define NH     16
#define DKH    64
#define MTOT   (BATCH * SEQ)   // 8192
#define NREL   (2 * SEQ - 1)   // 4095
#define WSPAD  2048            // f16 elements of slack between ws regions

static __device__ __forceinline__ v8f wmma_f16(v16h a, v16h b, v8f c) {
  return __builtin_amdgcn_wmma_f32_16x16x32_f16(false, a, false, b, (short)0, c,
                                                false, false);
}

// ---- fragment loaders -------------------------------------------------------
// A-matrix 16x32 layout: lanes 0-15 row=lane hold K 0..7 & 16..23; lanes 16-31
// same row hold K 8..15 & 24..31.
static __device__ __forceinline__ v16h fragA_f16(const _Float16* rowptr, int k) {
  const int koff = ((threadIdx.x & 31) >> 4) << 3;       // 0 or 8
  v8h lo = *(const v8h*)(rowptr + k + koff);
  v8h hi = *(const v8h*)(rowptr + k + koff + 16);
  v16h f;
#pragma unroll
  for (int i = 0; i < 8; ++i) { f[i] = lo[i]; f[i + 8] = hi[i]; }
  return f;
}
// B-matrix 32x16 layout: lane n<16 = column n holds K 0..7 & 16..23; lane n+16
// holds K 8..15 & 24..31 (same order as A).
static __device__ __forceinline__ v16h fragB_f16(const _Float16* colptr, int k) {
  const int koff = ((threadIdx.x & 31) >> 4) << 3;       // 0 or 8
  v8h lo = *(const v8h*)(colptr + k + koff);
  v8h hi = *(const v8h*)(colptr + k + koff + 16);
  v16h f;
#pragma unroll
  for (int i = 0; i < 8; ++i) { f[i] = lo[i]; f[i + 8] = hi[i]; }
  return f;
}

static __device__ __forceinline__ float redmax16(float v) {
  v = fmaxf(v, __shfl_xor(v, 1, 32));
  v = fmaxf(v, __shfl_xor(v, 2, 32));
  v = fmaxf(v, __shfl_xor(v, 4, 32));
  v = fmaxf(v, __shfl_xor(v, 8, 32));
  return v;
}
static __device__ __forceinline__ float redsum16(float v) {
  v += __shfl_xor(v, 1, 32);
  v += __shfl_xor(v, 2, 32);
  v += __shfl_xor(v, 4, 32);
  v += __shfl_xor(v, 8, 32);
  return v;
}

// ---- kernel 0a: bulk f32 -> f16 convert ------------------------------------
__global__ __launch_bounds__(256) void cvt_f32_f16_kernel(
    const float* __restrict__ src, _Float16* __restrict__ dst, int n4) {
  const int i = (blockIdx.x * 256 + threadIdx.x) * 4;
  if (i < n4) {
    float4 v = *(const float4*)(src + i);
    dst[i]     = (_Float16)v.x;
    dst[i + 1] = (_Float16)v.y;
    dst[i + 2] = (_Float16)v.z;
    dst[i + 3] = (_Float16)v.w;
  }
}

// ---- kernel 0b: rel-bias table transpose (NREL,H) -> (H,NREL) --------------
__global__ __launch_bounds__(256) void bias_transpose_kernel(
    const float* __restrict__ src, float* __restrict__ dst) {
  const int i = blockIdx.x * 256 + threadIdx.x;
  if (i < NREL * NH) {
    const int d = i / NH, h = i % NH;
    dst[h * NREL + d] = src[i];
  }
}

// ---- kernel 1: fused QKV projection (pipelined, all-f16) -------------------
// grid (MTOT/128, DMODEL/64, 3), block 128 (4 waves). Wave: 32x64 tile,
// one-iteration-ahead fragment prefetch -> WMMAs never wait loadcnt to 0.
__global__ __launch_bounds__(128) void qkv_proj_kernel(
    const _Float16* __restrict__ Xh, const _Float16* __restrict__ Wh,
    const float* __restrict__ b_q, const float* __restrict__ b_k,
    const float* __restrict__ b_v,
    _Float16* __restrict__ Qh, _Float16* __restrict__ Kh,
    _Float16* __restrict__ Vth) {
  const int wave   = threadIdx.x >> 5;
  const int lane   = threadIdx.x & 31;
  const int lane_n = lane & 15;
  const int z      = blockIdx.z;
  const int tileM  = (blockIdx.x * 4 + wave) * 32;
  const int tileN  = blockIdx.y * 64;

  const _Float16* X = Xh + (size_t)z * ((size_t)MTOT * DMODEL + WSPAD);
  const _Float16* W = Wh + (size_t)z * ((size_t)DMODEL * DMODEL + WSPAD);
  const float* bias = (z == 0) ? b_q : (z == 1) ? b_k : b_v;
  _Float16* outp    = (z == 0) ? Qh  : (z == 1) ? Kh  : Vth;

  const _Float16* arow0 = X + (size_t)(tileM + lane_n) * DMODEL;
  const _Float16* arow1 = arow0 + (size_t)16 * DMODEL;
  const _Float16* wcol0 = W + (size_t)(tileN + lane_n) * DMODEL;
  const _Float16* wcol1 = wcol0 + (size_t)16 * DMODEL;
  const _Float16* wcol2 = wcol0 + (size_t)32 * DMODEL;
  const _Float16* wcol3 = wcol0 + (size_t)48 * DMODEL;

  v8f acc[2][4] = {};
  v16h a0 = fragA_f16(arow0, 0);
  v16h a1 = fragA_f16(arow1, 0);
  v16h b0 = fragB_f16(wcol0, 0);
  v16h b1 = fragB_f16(wcol1, 0);
  v16h b2 = fragB_f16(wcol2, 0);
  v16h b3 = fragB_f16(wcol3, 0);
  for (int k = 0; k < DMODEL; k += 32) {
    const int kn = k + 32;  // last-iter prefetch lands in padded ws region
    v16h na0 = fragA_f16(arow0, kn);
    v16h na1 = fragA_f16(arow1, kn);
    v16h nb0 = fragB_f16(wcol0, kn);
    v16h nb1 = fragB_f16(wcol1, kn);
    v16h nb2 = fragB_f16(wcol2, kn);
    v16h nb3 = fragB_f16(wcol3, kn);
    acc[0][0] = wmma_f16(a0, b0, acc[0][0]);
    acc[1][0] = wmma_f16(a1, b0, acc[1][0]);
    acc[0][1] = wmma_f16(a0, b1, acc[0][1]);
    acc[1][1] = wmma_f16(a1, b1, acc[1][1]);
    acc[0][2] = wmma_f16(a0, b2, acc[0][2]);
    acc[1][2] = wmma_f16(a1, b2, acc[1][2]);
    acc[0][3] = wmma_f16(a0, b3, acc[0][3]);
    acc[1][3] = wmma_f16(a1, b3, acc[1][3]);
    a0 = na0; a1 = na1; b0 = nb0; b1 = nb1; b2 = nb2; b3 = nb3;
  }
  const int rowoff = (lane >> 4) << 3;
#pragma unroll
  for (int t = 0; t < 2; ++t) {
#pragma unroll
    for (int d = 0; d < 4; ++d) {
      const int   n  = tileN + d * 16 + lane_n;
      const int   h  = n >> 6, dk = n & 63;
      const float bv = bias[n];
#pragma unroll
      for (int r = 0; r < 8; ++r) {
        const int m  = tileM + t * 16 + r + rowoff;
        const int bb = m >> 11, s = m & (SEQ - 1);
        const float v = acc[t][d][r] + bv;
        size_t idx;
        if (z < 2)  // Q,K : (B,H,S,DK)
          idx = (((size_t)(bb * NH + h)) * SEQ + s) * DKH + dk;
        else        // V transposed : (B,H,DK,S)
          idx = (((size_t)(bb * NH + h)) * DKH + dk) * SEQ + s;
        outp[idx] = (_Float16)v;
      }
    }
  }
}

// ---- kernel 2: flash attention (K double-buffered, early V issue) ----------
// grid (SEQ/64, NH, BATCH), block 128 (4 waves); wave owns a 16-query tile,
// streams keys 32 at a time with online softmax.
__global__ __launch_bounds__(128) void flash_attn_kernel(
    const _Float16* __restrict__ Qh, const _Float16* __restrict__ Kh,
    const _Float16* __restrict__ Vth, const int* __restrict__ mask,
    const float* __restrict__ biasT, _Float16* __restrict__ Ctx) {
  __shared__ _Float16 plds[4][16 * 32];   // per-wave P staging (C->A relayout)
  const int wave   = threadIdx.x >> 5;
  const int lane   = threadIdx.x & 31;
  const int lane_n = lane & 15;
  const int rowoff = (lane >> 4) << 3;
  const int h      = blockIdx.y;
  const int bb     = blockIdx.z;
  const int qbase  = (blockIdx.x * 4 + wave) * 16;

  const _Float16* qhead = Qh  + ((size_t)(bb * NH + h)) * SEQ * DKH;
  const _Float16* khead = Kh  + ((size_t)(bb * NH + h)) * SEQ * DKH;
  const _Float16* vhead = Vth + ((size_t)(bb * NH + h)) * DKH * SEQ;
  const float*    bh    = biasT + (size_t)h * NREL;   // bh[q-k+2047], coalesced
  _Float16* P = plds[wave];

  const _Float16* qrow = qhead + (size_t)(qbase + lane_n) * DKH;
  const v16h aq0 = fragA_f16(qrow, 0);    // dk 0..31
  const v16h aq1 = fragA_f16(qrow, 32);   // dk 32..63

  // K fragment double-buffer: preload step 0
  const _Float16* krow0 = khead + (size_t)lane_n * DKH;
  const _Float16* krow1 = krow0 + (size_t)16 * DKH;
  v16h bk0 = fragB_f16(krow0, 0);
  v16h bk1 = fragB_f16(krow0, 32);
  v16h bk2 = fragB_f16(krow1, 0);
  v16h bk3 = fragB_f16(krow1, 32);

  v8f   o[4] = {};
  float mrow[8], lrow[8];
#pragma unroll
  for (int r = 0; r < 8; ++r) { mrow[r] = -1e30f; lrow[r] = 0.0f; }

  for (int ks = 0; ks < SEQ; ks += 32) {
    // issue this step's V loads early; consumed only after the softmax + LDS
    // relayout, so they overlap all of that latency.
    v16h bv0 = fragB_f16(vhead + (size_t)(lane_n)      * SEQ, ks);
    v16h bv1 = fragB_f16(vhead + (size_t)(16 + lane_n) * SEQ, ks);
    v16h bv2 = fragB_f16(vhead + (size_t)(32 + lane_n) * SEQ, ks);
    v16h bv3 = fragB_f16(vhead + (size_t)(48 + lane_n) * SEQ, ks);

    // scores: two 16x16 tiles, K-dim = DK = 64 -> 2 WMMAs each
    v8f c0 = {}, c1 = {};
    c0 = wmma_f16(aq0, bk0, c0);
    c0 = wmma_f16(aq1, bk1, c0);
    c1 = wmma_f16(aq0, bk2, c1);
    c1 = wmma_f16(aq1, bk3, c1);

    // prefetch next step's K fragments (last-iter read stays inside d_ws)
    {
      const _Float16* nk0 = khead + (size_t)(ks + 32 + lane_n) * DKH;
      const _Float16* nk1 = nk0 + (size_t)16 * DKH;
      bk0 = fragB_f16(nk0, 0);
      bk1 = fragB_f16(nk0, 32);
      bk2 = fragB_f16(nk1, 0);
      bk3 = fragB_f16(nk1, 32);
    }

#pragma unroll
    for (int r = 0; r < 8; ++r) {
      const int q  = qbase + r + rowoff;
      const int k0 = ks + lane_n;
      const int k1 = ks + 16 + lane_n;
      float s0 = c0[r] * 0.125f + bh[q - k0 + 2047];
      float s1 = c1[r] * 0.125f + bh[q - k1 + 2047];
      if (mask[(size_t)q * SEQ + k0] == 0) s0 = -1e9f;
      if (mask[(size_t)q * SEQ + k1] == 0) s1 = -1e9f;
      const float mcur = fmaxf(redmax16(s0), redmax16(s1));
      const float mnew = fmaxf(mrow[r], mcur);
      const float corr = __expf(mrow[r] - mnew);
      const float p0   = __expf(s0 - mnew);
      const float p1   = __expf(s1 - mnew);
      lrow[r] = lrow[r] * corr + redsum16(p0) + redsum16(p1);
      mrow[r] = mnew;
#pragma unroll
      for (int d = 0; d < 4; ++d) o[d][r] *= corr;
      P[(r + rowoff) * 32 + lane_n]      = (_Float16)p0;
      P[(r + rowoff) * 32 + 16 + lane_n] = (_Float16)p1;
    }
    __builtin_amdgcn_wave_barrier();      // DS ops in-order per wave; block reordering
    const v16h pa = fragA_f16(P + lane_n * 32, 0);
    o[0] = wmma_f16(pa, bv0, o[0]);
    o[1] = wmma_f16(pa, bv1, o[1]);
    o[2] = wmma_f16(pa, bv2, o[2]);
    o[3] = wmma_f16(pa, bv3, o[3]);
    __builtin_amdgcn_wave_barrier();
  }
#pragma unroll
  for (int r = 0; r < 8; ++r) {
    const float inv = 1.0f / lrow[r];
    const int   s   = qbase + r + rowoff;
#pragma unroll
    for (int d = 0; d < 4; ++d) {
      const int col = h * DKH + d * 16 + lane_n;
      Ctx[((size_t)(bb * SEQ + s)) * DMODEL + col] = (_Float16)(o[d][r] * inv);
    }
  }
}

// ---- kernel 3: output projection (pipelined, all-f16, f32 out) -------------
__global__ __launch_bounds__(128) void out_proj_kernel(
    const _Float16* __restrict__ Ctx, const _Float16* __restrict__ Woh,
    const float* __restrict__ b_o, float* __restrict__ out) {
  const int wave   = threadIdx.x >> 5;
  const int lane   = threadIdx.x & 31;
  const int lane_n = lane & 15;
  const int tileM  = (blockIdx.x * 4 + wave) * 32;
  const int tileN  = blockIdx.y * 64;

  const _Float16* arow0 = Ctx + (size_t)(tileM + lane_n) * DMODEL;
  const _Float16* arow1 = arow0 + (size_t)16 * DMODEL;
  const _Float16* wcol0 = Woh + (size_t)(tileN + lane_n) * DMODEL;
  const _Float16* wcol1 = wcol0 + (size_t)16 * DMODEL;
  const _Float16* wcol2 = wcol0 + (size_t)32 * DMODEL;
  const _Float16* wcol3 = wcol0 + (size_t)48 * DMODEL;

  v8f acc[2][4] = {};
  v16h a0 = fragA_f16(arow0, 0);
  v16h a1 = fragA_f16(arow1, 0);
  v16h b0 = fragB_f16(wcol0, 0);
  v16h b1 = fragB_f16(wcol1, 0);
  v16h b2 = fragB_f16(wcol2, 0);
  v16h b3 = fragB_f16(wcol3, 0);
  for (int k = 0; k < DMODEL; k += 32) {
    const int kn = k + 32;  // last-iter prefetch lands in padded ws region
    v16h na0 = fragA_f16(arow0, kn);
    v16h na1 = fragA_f16(arow1, kn);
    v16h nb0 = fragB_f16(wcol0, kn);
    v16h nb1 = fragB_f16(wcol1, kn);
    v16h nb2 = fragB_f16(wcol2, kn);
    v16h nb3 = fragB_f16(wcol3, kn);
    acc[0][0] = wmma_f16(a0, b0, acc[0][0]);
    acc[1][0] = wmma_f16(a1, b0, acc[1][0]);
    acc[0][1] = wmma_f16(a0, b1, acc[0][1]);
    acc[1][1] = wmma_f16(a1, b1, acc[1][1]);
    acc[0][2] = wmma_f16(a0, b2, acc[0][2]);
    acc[1][2] = wmma_f16(a1, b2, acc[1][2]);
    acc[0][3] = wmma_f16(a0, b3, acc[0][3]);
    acc[1][3] = wmma_f16(a1, b3, acc[1][3]);
    a0 = na0; a1 = na1; b0 = nb0; b1 = nb1; b2 = nb2; b3 = nb3;
  }
  const int rowoff = (lane >> 4) << 3;
#pragma unroll
  for (int t = 0; t < 2; ++t) {
#pragma unroll
    for (int d = 0; d < 4; ++d) {
      const int   n  = tileN + d * 16 + lane_n;
      const float bv = b_o[n];
#pragma unroll
      for (int r = 0; r < 8; ++r) {
        const int m = tileM + t * 16 + r + rowoff;
        out[(size_t)m * DMODEL + n] = acc[t][d][r] + bv;
      }
    }
  }
}

extern "C" void kernel_launch(void* const* d_in, const int* in_sizes, int n_in,
                              void* d_out, int out_size, void* d_ws,
                              size_t ws_size, hipStream_t stream) {
  const float* query = (const float*)d_in[0];
  const float* key   = (const float*)d_in[1];
  const float* value = (const float*)d_in[2];
  const int*   mask  = (const int*)d_in[3];
  const float* w_q   = (const float*)d_in[4];
  const float* b_q   = (const float*)d_in[5];
  const float* w_k   = (const float*)d_in[6];
  const float* b_k   = (const float*)d_in[7];
  const float* w_v   = (const float*)d_in[8];
  const float* b_v   = (const float*)d_in[9];
  const float* w_o   = (const float*)d_in[10];
  const float* b_o   = (const float*)d_in[11];
  const float* reb   = (const float*)d_in[12];

  // workspace layout (each region padded by WSPAD f16 for prefetch overrun)
  const size_t nX   = (size_t)MTOT * DMODEL;           // 8,388,608 per tensor
  const size_t nW   = (size_t)DMODEL * DMODEL;         // 1,048,576 per weight
  const size_t nQKV = (size_t)BATCH * NH * SEQ * DKH;  // 8,388,608

  float*    biasT = (float*)d_ws;                      // (H, NREL) f32
  _Float16* Xh    = (_Float16*)(biasT + (size_t)NH * NREL + 16);
  _Float16* Wh    = Xh + 3 * (nX + WSPAD);             // w_q|w_k|w_v|w_o f16
  _Float16* Qh    = Wh + 4 * (nW + WSPAD);
  _Float16* Kh    = Qh + nQKV + WSPAD;
  _Float16* Vth   = Kh + nQKV + WSPAD;
  _Float16* Ctx   = Vth + nQKV + WSPAD;                // + nX + WSPAD total

  // phase 0: precision conversion + bias transpose
  const int gX = (int)(nX / 4 / 256), gW = (int)(nW / 4 / 256);
  cvt_f32_f16_kernel<<<gX, 256, 0, stream>>>(query, Xh,                  (int)nX);
  cvt_f32_f16_kernel<<<gX, 256, 0, stream>>>(key,   Xh + (nX + WSPAD),   (int)nX);
  cvt_f32_f16_kernel<<<gX, 256, 0, stream>>>(value, Xh + 2 * (nX + WSPAD), (int)nX);
  cvt_f32_f16_kernel<<<gW, 256, 0, stream>>>(w_q, Wh,                  (int)nW);
  cvt_f32_f16_kernel<<<gW, 256, 0, stream>>>(w_k, Wh + (nW + WSPAD),   (int)nW);
  cvt_f32_f16_kernel<<<gW, 256, 0, stream>>>(w_v, Wh + 2 * (nW + WSPAD), (int)nW);
  cvt_f32_f16_kernel<<<gW, 256, 0, stream>>>(w_o, Wh + 3 * (nW + WSPAD), (int)nW);
  bias_transpose_kernel<<<(NREL * NH + 255) / 256, 256, 0, stream>>>(reb, biasT);

  const dim3 blk(128);
  qkv_proj_kernel<<<dim3(MTOT / 128, DMODEL / 64, 3), blk, 0, stream>>>(
      Xh, Wh, b_q, b_k, b_v, Qh, Kh, Vth);
  flash_attn_kernel<<<dim3(SEQ / 64, NH, BATCH), blk, 0, stream>>>(
      Qh, Kh, Vth, mask, biasT, Ctx);
  out_proj_kernel<<<dim3(MTOT / 128, DMODEL / 64), blk, 0, stream>>>(
      Ctx, Wh + 3 * (nW + WSPAD), b_o, (float*)d_out);
}
